// DiagramAnalyzer_78125455114747
// MI455X (gfx1250) — compile-verified, matrix operand order checked by reading, not run on hardware
//
#include <hip/hip_runtime.h>

typedef __attribute__((ext_vector_type(16))) _Float16 v16h;
typedef __attribute__((ext_vector_type(8)))  float    v8f;

// Tensor Data Mover availability (arity differs across toolchains; see
// CDNA5_HIP.md: ROCm 7.2 clang-22 = 5 args, amdgpu-toolchain clang-23 = 6).
#if defined(__has_builtin)
#if __has_builtin(__builtin_amdgcn_tensor_load_to_lds)
#define USE_TDM 1
#endif
#endif
#ifndef USE_TDM
#define USE_TDM 0
#endif

#if USE_TDM
typedef __attribute__((ext_vector_type(4))) unsigned int v4u;
typedef __attribute__((ext_vector_type(8))) int          v8i_t;
typedef __attribute__((ext_vector_type(4))) int          v4i_t;
#endif

#define EPSV 1e-5f

// ---------------------------------------------------------------------------
// WMMA fragment loader (CDNA5 wave32).
// A (16x32 f16) and B (32x16 f16) fragments share the same per-lane pattern:
// lane L holds A-row / B-col (L & 15); its 16 halves cover
//   K = (v/4)*16 + (L<16 ? 0 : 8) + 2*(v%4) , +1      for v = 0..7
// i.e. two contiguous 16-byte runs, which lower to two ds_load_b128.
// ---------------------------------------------------------------------------
__device__ __forceinline__ v16h load_frag(const _Float16* __restrict__ lds,
                                          int row, int stride, int lane)
{
  v16h f;
  const int b8 = (lane < 16) ? 0 : 8;
#pragma unroll
  for (int v = 0; v < 8; ++v) {
    const int k = ((v >> 2) << 4) + b8 + ((v & 3) << 1);
    union { unsigned u; _Float16 h[2]; } cv;
    cv.u = *reinterpret_cast<const unsigned*>(lds + row * stride + k);
    f[2 * v]     = cv.h[0];
    f[2 * v + 1] = cv.h[1];
  }
  return f;
}

// ---------------------------------------------------------------------------
// Generic GEMM: C[M,N] = act(A[M,K] @ W[N,K]^T + bias), f16 WMMA, f32 accum.
// Block: 256 threads = 8 waves, tile 128(M) x 64(N), K stepped by 32.
// Wave grid 4(M) x 2(N); each wave owns a 32x32 tile -> 4 WMMAs per K-step
// reusing 2 A-fragments x 2 B-fragments (2 LDS loads per WMMA).
// ---------------------------------------------------------------------------
__global__ __launch_bounds__(256) void gemm_wmma(
    const float* __restrict__ A, int lda,
    const float* __restrict__ W, int ldw,
    const float* __restrict__ bias,
    float* __restrict__ C, int ldc,
    int M, int N, int K, int act)
{
  __shared__ _Float16 As[128 * 40];
  __shared__ _Float16 Bs[64 * 40];
  const int tid  = threadIdx.x;
  const int lane = tid & 31;
  const int wave = tid >> 5;
  const int wm   = wave & 3;   // 4 waves along M
  const int wn   = wave >> 2;  // 2 waves along N
  const int m0   = blockIdx.y * 128;
  const int n0   = blockIdx.x * 64;

  v8f acc00 = {}, acc01 = {}, acc10 = {}, acc11 = {};
  const int r  = tid >> 1;          // staging row
  const int ch = (tid & 1) * 16;    // staging half-row (16 elements)

  for (int k0 = 0; k0 < K; k0 += 32) {
    // ---- stage A tile: 128 rows x 32 (all 256 threads, 16 f32 each) ----
    {
      const int m = m0 + r;
      _Float16* dst = As + r * 40 + ch;
      if (m < M) {
        const float* ap = A + (size_t)m * lda + k0 + ch;
#pragma unroll
        for (int j = 0; j < 16; j += 4) {
          float4 f = *reinterpret_cast<const float4*>(ap + j);
          dst[j + 0] = (_Float16)f.x; dst[j + 1] = (_Float16)f.y;
          dst[j + 2] = (_Float16)f.z; dst[j + 3] = (_Float16)f.w;
        }
        if (k0 + 32 < K) __builtin_prefetch(ap + 32, 0, 3);
      } else {
#pragma unroll
        for (int j = 0; j < 16; ++j) dst[j] = (_Float16)0.f;
      }
    }
    // ---- stage B tile: 64 rows x 32 (threads 0..127, 16 f32 each) ----
    if (tid < 128) {
      const int n = n0 + r;
      _Float16* dst = Bs + r * 40 + ch;
      if (n < N) {
        const float* wp = W + (size_t)n * ldw + k0 + ch;
#pragma unroll
        for (int j = 0; j < 16; j += 4) {
          float4 f = *reinterpret_cast<const float4*>(wp + j);
          dst[j + 0] = (_Float16)f.x; dst[j + 1] = (_Float16)f.y;
          dst[j + 2] = (_Float16)f.z; dst[j + 3] = (_Float16)f.w;
        }
        if (k0 + 32 < K) __builtin_prefetch(wp + 32, 0, 3);
      } else {
#pragma unroll
        for (int j = 0; j < 16; ++j) dst[j] = (_Float16)0.f;
      }
    }
    __syncthreads();
    const int ar = wm * 32 + (lane & 15);
    const int br = wn * 32 + (lane & 15);
    v16h a0 = load_frag(As, ar,      40, lane);
    v16h a1 = load_frag(As, ar + 16, 40, lane);
    v16h b0 = load_frag(Bs, br,      40, lane);
    v16h b1 = load_frag(Bs, br + 16, 40, lane);
    acc00 = __builtin_amdgcn_wmma_f32_16x16x32_f16(false, a0, false, b0,
                                                   (short)0, acc00, false, false);
    acc01 = __builtin_amdgcn_wmma_f32_16x16x32_f16(false, a0, false, b1,
                                                   (short)0, acc01, false, false);
    acc10 = __builtin_amdgcn_wmma_f32_16x16x32_f16(false, a1, false, b0,
                                                   (short)0, acc10, false, false);
    acc11 = __builtin_amdgcn_wmma_f32_16x16x32_f16(false, a1, false, b1,
                                                   (short)0, acc11, false, false);
    __syncthreads();
  }

  // ---- epilogue ----
  const int lm = (lane < 16) ? 0 : 8;
  const int ln = lane & 15;
#pragma unroll
  for (int fm = 0; fm < 2; ++fm) {
#pragma unroll
    for (int fn = 0; fn < 2; ++fn) {
      const v8f accv = (fm == 0) ? (fn == 0 ? acc00 : acc01)
                                 : (fn == 0 ? acc10 : acc11);
      const int n = n0 + wn * 32 + fn * 16 + ln;
      if (n < N) {
        const float bv = bias ? bias[n] : 0.f;
#pragma unroll
        for (int i = 0; i < 8; ++i) {
          const int m = m0 + wm * 32 + fm * 16 + i + lm;
          if (m < M) {
            float rr = accv[i] + bv;
            if (act == 1)      rr = fmaxf(rr, 0.f);
            else if (act == 2) rr = 1.f / (1.f + __expf(-rr));
            C[(size_t)m * ldc + n] = rr;
          }
        }
      }
    }
  }
}

// ---------------------------------------------------------------------------
// Implicit-GEMM 3x3 conv, stride 1 pad 1, NHWC 64x64 images.
// A rows = output pixels (tap-major K: each 32-wide K-step is one contiguous
// 32-channel slice at a shifted position), B rows = f16 weights [Cout][9][C].
// The B panel (64 rows x 32 halves, row stride 9*C halves) is DMA'd into LDS
// by the Tensor Data Mover when available: a 2D D# tile with LDS padding
// (pad_interval = 16 DWORDs, pad_amount = 4 DWORDs) reproduces the 40-half
// LDS row pitch used by the WMMA fragment loads.
// ---------------------------------------------------------------------------
__global__ __launch_bounds__(256) void conv3x3_wmma(
    const float* __restrict__ in,      // [B][64][64][C]
    const _Float16* __restrict__ wq,   // [Cout][9][C]
    const float* __restrict__ cb,
    const float* __restrict__ g,  const float* __restrict__ bb,
    const float* __restrict__ rm, const float* __restrict__ rv,
    float* __restrict__ out,           // [B][4096][Cout]
    int C, int Cout, int fuse_bn)
{
  __shared__ _Float16 As[128 * 40];
  __shared__ _Float16 Bs[64 * 40];
  const int tid  = threadIdx.x;
  const int lane = tid & 31;
  const int wave = tid >> 5;
  const int wm = wave & 3, wn = wave >> 2;
  const int b  = blockIdx.z;
  const int m0 = blockIdx.y * 128;    // pixel tile
  const int n0 = blockIdx.x * 64;     // out-channel tile
  const int K  = 9 * C;

  v8f acc00 = {}, acc01 = {}, acc10 = {}, acc11 = {};
  const int r  = tid >> 1;
  const int ch = (tid & 1) * 16;

  for (int k0 = 0; k0 < K; k0 += 32) {
    const int tap = k0 / C;
    const int c0  = k0 - tap * C;
    const int dy = tap / 3 - 1, dx = tap % 3 - 1;
    // ---- stage A: 128 pixels x 32 channels ----
    {
      const int pix = m0 + r;
      const int y = pix >> 6, x = pix & 63;
      const int ys = y + dy, xs = x + dx;
      _Float16* dst = As + r * 40 + ch;
      if ((unsigned)ys < 64u && (unsigned)xs < 64u) {
        const float* ap = in + (((size_t)b * 4096 + (ys * 64 + xs)) * C + c0 + ch);
#pragma unroll
        for (int j = 0; j < 16; j += 4) {
          float4 f = *reinterpret_cast<const float4*>(ap + j);
          dst[j + 0] = (_Float16)f.x; dst[j + 1] = (_Float16)f.y;
          dst[j + 2] = (_Float16)f.z; dst[j + 3] = (_Float16)f.w;
        }
      } else {
#pragma unroll
        for (int j = 0; j < 16; ++j) dst[j] = (_Float16)0.f;
      }
    }
#if USE_TDM
    // ---- stage B via Tensor Data Mover (wave 0 issues; all waves sync) ----
    if (wave == 0) {
      const unsigned long long ga = (unsigned long long)
          (const void*)(wq + (((size_t)n0 * 9 + tap) * C + c0));
      // generic LDS pointers carry the LDS byte offset in the low 32 bits
      const unsigned lds_addr =
          (unsigned)(unsigned long long)(const void*)&Bs[0];
      v4u g0;
      g0[0] = 1u;                                   // count=1, user descriptor
      g0[1] = lds_addr;                             // D#.lds_addr
      g0[2] = (unsigned)ga;                         // global_addr[31:0]
      g0[3] = (unsigned)((ga >> 32) & 0x01FFFFFFull) | (2u << 30); // type=2
      v8i_t g1;
      g1[0] = (int)((1u << 16)     // data_size = 2 bytes
                  | (1u << 20)     // pad_enable (LDS row pitch padding)
                  | (3u << 22)     // pad_interval: 16 DWORDs (one 32-half row)
                  | (3u << 25));   // pad_amount:   4 DWORDs (8 halves)
      g1[1] = (int)(32u << 16);    // tensor_dim0 = 32 elements
      g1[2] = (int)(64u << 16);    // tensor_dim1 = 64 rows
      g1[3] = (int)(32u << 16);    // tile_dim0 = 32
      g1[4] = 64;                  // tile_dim1 = 64
      g1[5] = 9 * C;               // tensor_dim0_stride (elements)
      g1[6] = 0;
      g1[7] = 0;
      v4i_t gz = {0, 0, 0, 0};
#if defined(__clang_major__) && __clang_major__ >= 23
      v8i_t gz8 = {0, 0, 0, 0, 0, 0, 0, 0};
      __builtin_amdgcn_tensor_load_to_lds(g0, g1, gz, gz, gz8, 0);
#else
      __builtin_amdgcn_tensor_load_to_lds(g0, g1, gz, gz, 0);
#endif
      __builtin_amdgcn_s_wait_tensorcnt(0);
    }
#else
    // ---- stage B: 64 out-channels x 32 (already f16, contiguous) ----
    if (tid < 128) {
      const int n = n0 + r;            // always < Cout (Cout % 64 == 0)
      const _Float16* wp = wq + (((size_t)n * 9 + tap) * C + c0 + ch);
      *reinterpret_cast<uint4*>(&Bs[r * 40 + ch]) =
          *reinterpret_cast<const uint4*>(wp);
      *reinterpret_cast<uint4*>(&Bs[r * 40 + ch + 8]) =
          *reinterpret_cast<const uint4*>(wp + 8);
      if (k0 + 32 < K) __builtin_prefetch(wp + 32, 0, 3);
    }
#endif
    __syncthreads();
    const int ar = wm * 32 + (lane & 15);
    const int br = wn * 32 + (lane & 15);
    v16h a0 = load_frag(As, ar,      40, lane);
    v16h a1 = load_frag(As, ar + 16, 40, lane);
    v16h b0 = load_frag(Bs, br,      40, lane);
    v16h b1 = load_frag(Bs, br + 16, 40, lane);
    acc00 = __builtin_amdgcn_wmma_f32_16x16x32_f16(false, a0, false, b0,
                                                   (short)0, acc00, false, false);
    acc01 = __builtin_amdgcn_wmma_f32_16x16x32_f16(false, a0, false, b1,
                                                   (short)0, acc01, false, false);
    acc10 = __builtin_amdgcn_wmma_f32_16x16x32_f16(false, a1, false, b0,
                                                   (short)0, acc10, false, false);
    acc11 = __builtin_amdgcn_wmma_f32_16x16x32_f16(false, a1, false, b1,
                                                   (short)0, acc11, false, false);
    __syncthreads();
  }

  const int lm = (lane < 16) ? 0 : 8;
  const int ln = lane & 15;
#pragma unroll
  for (int fn = 0; fn < 2; ++fn) {
    const int n = n0 + wn * 32 + fn * 16 + ln;
    float scale = 1.f, shift = 0.f;
    if (fuse_bn) {
      scale = g[n] * rsqrtf(rv[n] + EPSV);
      shift = bb[n] - rm[n] * scale;
    }
    const float bv = cb[n];
#pragma unroll
    for (int fm = 0; fm < 2; ++fm) {
      const v8f accv = (fm == 0) ? (fn == 0 ? acc00 : acc01)
                                 : (fn == 0 ? acc10 : acc11);
#pragma unroll
      for (int i = 0; i < 8; ++i) {
        const int m = m0 + wm * 32 + fm * 16 + i + lm;
        float rr = accv[i] + bv;
        if (fuse_bn) rr = fmaxf(rr * scale + shift, 0.f);
        out[((size_t)b * 4096 + m) * Cout + n] = rr;
      }
    }
  }
}

// ---------------------------------------------------------------------------
// conv1: 7x7 stride 2 pad 3, NCHW in [8,3,256,256] -> NHWC out [8,128,128,64],
// fused BN + ReLU (only 3 input channels: direct scalar conv is fine).
// ---------------------------------------------------------------------------
__global__ __launch_bounds__(256) void conv1_bn_relu(
    const float* __restrict__ in, const float* __restrict__ w,
    const float* __restrict__ cb, const float* __restrict__ g,
    const float* __restrict__ bb, const float* __restrict__ rm,
    const float* __restrict__ rv, float* __restrict__ out)
{
  const int idx = blockIdx.x * 256 + threadIdx.x;   // 8*128*128*64 = 2^23
  const int co = idx & 63, x = (idx >> 6) & 127, y = (idx >> 13) & 127,
            b = idx >> 20;
  float acc = cb[co];
  for (int ci = 0; ci < 3; ++ci)
    for (int ky = 0; ky < 7; ++ky) {
      const int iy = 2 * y - 3 + ky;
      if ((unsigned)iy >= 256u) continue;
      for (int kx = 0; kx < 7; ++kx) {
        const int ix = 2 * x - 3 + kx;
        if ((unsigned)ix >= 256u) continue;
        acc += in[((b * 3 + ci) * 256 + iy) * 256 + ix] *
               w[((co * 3 + ci) * 7 + ky) * 7 + kx];
      }
    }
  const float sc = g[co] * rsqrtf(rv[co] + EPSV);
  out[idx] = fmaxf((acc - rm[co]) * sc + bb[co], 0.f);
}

// 3x3 stride-2 pad-1 maxpool, NHWC [8,128,128,64] -> [8,64,64,64]
__global__ __launch_bounds__(256) void maxpool_kernel(
    const float* __restrict__ in, float* __restrict__ out)
{
  const int idx = blockIdx.x * 256 + threadIdx.x;   // 2^21
  const int c = idx & 63, x = (idx >> 6) & 63, y = (idx >> 12) & 63,
            b = idx >> 18;
  float m = -3.4e38f;
  for (int ky = 0; ky < 3; ++ky) {
    const int iy = 2 * y - 1 + ky;
    if ((unsigned)iy >= 128u) continue;
    for (int kx = 0; kx < 3; ++kx) {
      const int ix = 2 * x - 1 + kx;
      if ((unsigned)ix >= 128u) continue;
      m = fmaxf(m, in[((b * 128 + iy) * 128 + ix) * 64 + c]);
    }
  }
  out[idx] = m;
}

// repack OIHW f32 [Cout][Cin][3][3] -> f16 [Cout][9][Cin]
__global__ __launch_bounds__(256) void repack_w3x3(
    const float* __restrict__ w, _Float16* __restrict__ wq, int Cout, int Cin)
{
  const int idx = blockIdx.x * 256 + threadIdx.x;
  const int total = Cout * Cin * 9;
  if (idx >= total) return;
  const int t = idx % 9;
  const int c = (idx / 9) % Cin;
  const int n = idx / (9 * Cin);
  wq[((size_t)n * 9 + t) * Cin + c] = (_Float16)w[idx];
}

// tgt[b,i,d] = obj_q[i,d] + pos_enc[0,i,d]
__global__ __launch_bounds__(256) void init_tgt_kernel(
    const float* __restrict__ objq, const float* __restrict__ pos,
    float* __restrict__ tgt)
{
  const int idx = blockIdx.x * 256 + threadIdx.x;   // 8*100*256
  const int qd = idx % 25600;
  tgt[idx] = objq[qd] + pos[qd];
}

// ---------------------------------------------------------------------------
// Softmax attention, one block per (b, head, query). dh = 32.
// ---------------------------------------------------------------------------
__global__ __launch_bounds__(256) void attention_kernel(
    const float* __restrict__ Q, int q_stride,
    const float* __restrict__ Kp, const float* __restrict__ Vp, int kv_stride,
    int kv_len, float* __restrict__ O, int nq, float scale)
{
  const int b = blockIdx.z, h = blockIdx.y, qi = blockIdx.x;
  const int tid = threadIdx.x;
  __shared__ float qsh[32];
  __shared__ float red[256];
  __shared__ float opart[32 * 256];
  const float* qrow = Q + (size_t)(b * nq + qi) * q_stride + h * 32;
  if (tid < 32) qsh[tid] = qrow[tid];
  __syncthreads();

  float mx = -3.0e38f;
  for (int j = tid; j < kv_len; j += 256) {
    const float* kr = Kp + (size_t)(b * kv_len + j) * kv_stride + h * 32;
    float s = 0.f;
#pragma unroll
    for (int d = 0; d < 32; ++d) s += qsh[d] * kr[d];
    mx = fmaxf(mx, s * scale);
  }
  red[tid] = mx; __syncthreads();
  for (int s = 128; s > 0; s >>= 1) {
    if (tid < s) red[tid] = fmaxf(red[tid], red[tid + s]);
    __syncthreads();
  }
  const float gmax = red[0]; __syncthreads();

  float accv[32];
#pragma unroll
  for (int d = 0; d < 32; ++d) accv[d] = 0.f;
  float se = 0.f;
  for (int j = tid; j < kv_len; j += 256) {
    const float* kr = Kp + (size_t)(b * kv_len + j) * kv_stride + h * 32;
    float s = 0.f;
#pragma unroll
    for (int d = 0; d < 32; ++d) s += qsh[d] * kr[d];
    const float e = __expf(s * scale - gmax);
    se += e;
    const float* vr = Vp + (size_t)(b * kv_len + j) * kv_stride + h * 32;
#pragma unroll
    for (int d = 0; d < 32; ++d) accv[d] += e * vr[d];
  }
  red[tid] = se; __syncthreads();
  for (int s = 128; s > 0; s >>= 1) {
    if (tid < s) red[tid] += red[tid + s];
    __syncthreads();
  }
  const float tot = red[0];
#pragma unroll
  for (int d = 0; d < 32; ++d) opart[d * 256 + tid] = accv[d];
  __syncthreads();
  if (tid < 32) {
    float s = 0.f;
    for (int t = 0; t < 256; ++t) s += opart[tid * 256 + t];
    O[(size_t)(b * nq + qi) * 256 + h * 32 + tid] = s / tot;
  }
}

// out = LayerNorm(a + r) * g + b   (D = 256, one block per row)
__global__ __launch_bounds__(256) void add_ln_kernel(
    const float* __restrict__ a, const float* __restrict__ r,
    const float* __restrict__ g, const float* __restrict__ bta,
    float* __restrict__ out)
{
  const int row = blockIdx.x, tid = threadIdx.x;
  __shared__ float red[256];
  const float v = a[(size_t)row * 256 + tid] + r[(size_t)row * 256 + tid];
  red[tid] = v; __syncthreads();
  for (int s = 128; s > 0; s >>= 1) {
    if (tid < s) red[tid] += red[tid + s];
    __syncthreads();
  }
  const float mean = red[0] * (1.f / 256.f); __syncthreads();
  const float d = v - mean;
  red[tid] = d * d; __syncthreads();
  for (int s = 128; s > 0; s >>= 1) {
    if (tid < s) red[tid] += red[tid + s];
    __syncthreads();
  }
  const float var = red[0] * (1.f / 256.f);
  out[(size_t)row * 256 + tid] = d * rsqrtf(var + EPSV) * g[tid] + bta[tid];
}

// softmax over 50 classes (one block of 64 per row)
__global__ __launch_bounds__(64) void softmax50_kernel(
    const float* __restrict__ in, float* __restrict__ out)
{
  const int row = blockIdx.x, tid = threadIdx.x;
  __shared__ float red[64];
  const float v = (tid < 50) ? in[(size_t)row * 50 + tid] : -3.0e38f;
  red[tid] = v; __syncthreads();
  for (int s = 32; s > 0; s >>= 1) {
    if (tid < s) red[tid] = fmaxf(red[tid], red[tid + s]);
    __syncthreads();
  }
  const float mx = red[0]; __syncthreads();
  const float e = (tid < 50) ? __expf(v - mx) : 0.f;
  red[tid] = e; __syncthreads();
  for (int s = 32; s > 0; s >>= 1) {
    if (tid < s) red[tid] += red[tid + s];
    __syncthreads();
  }
  if (tid < 50) out[(size_t)row * 50 + tid] = e / red[0];
}

// relations[b,p,:] = relu(U[b,i]+V[b,j]) @ w2^T + b2 ; p -> (i,j), i != j
__global__ __launch_bounds__(256) void relations_kernel(
    const float* __restrict__ U, const float* __restrict__ V,
    const float* __restrict__ w2, const float* __restrict__ b2,
    float* __restrict__ out)
{
  const int p = blockIdx.x, b = blockIdx.y, tid = threadIdx.x;
  const int i = p / 99, rr = p % 99;
  const int j = rr + (rr >= i ? 1 : 0);
  __shared__ float hbuf[256];
  const float h = U[(size_t)(b * 100 + i) * 256 + tid] +
                  V[(size_t)(b * 100 + j) * 256 + tid];
  hbuf[tid] = fmaxf(h, 0.f);
  __syncthreads();
  if (tid < 16) {
    const float* wr = w2 + tid * 256;
    float s = b2[tid];
    for (int d = 0; d < 256; ++d) s += hbuf[d] * wr[d];
    out[((size_t)b * 9900 + p) * 16 + tid] = s;
  }
}

// ---------------------------------------------------------------------------
// Host orchestration
// ---------------------------------------------------------------------------
extern "C" void kernel_launch(void* const* d_in, const int* in_sizes, int n_in,
                              void* d_out, int out_size, void* d_ws,
                              size_t ws_size, hipStream_t stream)
{
  (void)in_sizes; (void)n_in; (void)out_size; (void)ws_size;
  auto F = [&](int i) { return (const float*)d_in[i]; };
  float* ws  = (float*)d_ws;
  float* out = (float*)d_out;

  // workspace layout (float offsets); decoder scratch aliases dead conv bufs
  float* x1   = ws + 0;           // 8*128*128*64
  float* xp   = ws + 8388608;     // 8*64*64*64
  float* x2   = ws + 10485760;    // 8*64*64*128
  float* x3   = ws + 14680064;    // 8*64*64*256
  float* kv   = ws + 0;           // 8*4096*512   (aliases x1/xp/x2)
  float* qkv  = ws + 16777216;    // 800*768
  float* qb   = ws + 17391616;    // 800*256
  float* attn = ws + 17596416;    // 800*256
  float* proj = ws + 17801216;    // 800*256
  float* ff1  = ws + 18006016;    // 800*1024
  float* ff2  = ws + 18825216;    // 800*256
  float* Ub   = ws + 19030016;    // 800*256
  float* Vb   = ws + 19234816;    // 800*256
  float* tmp  = ws + 19439616;    // 800*256
  float* mem  = ws + 23068672;    // 8*4096*256 (persistent)
  float* tgt  = ws + 31457280;    // 8*100*256  (persistent)
  _Float16* w2q = (_Float16*)(ws + 31662080);  // 128*9*64
  _Float16* w3q = (_Float16*)(ws + 31698944);  // 256*9*128
  _Float16* w4q = (_Float16*)(ws + 31846400);  // 256*9*256

  // output layout (flat, return order)
  float* o_cls  = out;
  float* o_prob = out + 40000;
  float* o_bbox = out + 80000;
  float* o_lab  = out + 83200;
  float* o_rel  = out + 8083200;
  float* o_dec  = out + 9350400;

  auto gemm = [&](const float* A, int lda, const float* W, int ldw,
                  const float* bias, float* Cc, int ldc,
                  int M, int N, int K, int act) {
    dim3 grid((N + 63) / 64, (M + 127) / 128);
    gemm_wmma<<<grid, 256, 0, stream>>>(A, lda, W, ldw, bias, Cc, ldc,
                                        M, N, K, act);
  };

  // ---- backbone ----
  conv1_bn_relu<<<32768, 256, 0, stream>>>(F(0), F(1), F(2), F(3), F(4),
                                           F(5), F(6), x1);
  maxpool_kernel<<<8192, 256, 0, stream>>>(x1, xp);
  repack_w3x3<<<(128 * 64 * 9 + 255) / 256, 256, 0, stream>>>(F(7), w2q, 128, 64);
  repack_w3x3<<<(256 * 128 * 9 + 255) / 256, 256, 0, stream>>>(F(13), w3q, 256, 128);
  repack_w3x3<<<(256 * 256 * 9 + 255) / 256, 256, 0, stream>>>(F(19), w4q, 256, 256);
  conv3x3_wmma<<<dim3(2, 32, 8), 256, 0, stream>>>(
      xp, w2q, F(8), F(9), F(10), F(11), F(12), x2, 64, 128, 1);
  conv3x3_wmma<<<dim3(4, 32, 8), 256, 0, stream>>>(
      x2, w3q, F(14), F(15), F(16), F(17), F(18), x3, 128, 256, 1);
  conv3x3_wmma<<<dim3(4, 32, 8), 256, 0, stream>>>(
      x3, w4q, F(20), nullptr, nullptr, nullptr, nullptr, mem, 256, 256, 0);
  // mem (NHWC conv4 output) == [B, 4096, D] memory directly.

  // ---- decoder ----
  init_tgt_kernel<<<800, 256, 0, stream>>>(F(22), F(21), tgt);
  const float scale_att = 0.17677669529663689f;  // 1/sqrt(32)
  for (int l = 0; l < 4; ++l) {
    const int B0 = 23 + 18 * l;
    // self-attention
    gemm(tgt, 256, F(B0 + 0), 256, F(B0 + 1), qkv, 768, 800, 768, 256, 0);
    attention_kernel<<<dim3(100, 8, 8), 256, 0, stream>>>(
        qkv, 768, qkv + 256, qkv + 512, 768, 100, attn, 100, scale_att);
    gemm(attn, 256, F(B0 + 2), 256, F(B0 + 3), proj, 256, 800, 256, 256, 0);
    add_ln_kernel<<<800, 256, 0, stream>>>(tgt, proj, F(B0 + 12), F(B0 + 13), tgt);
    // cross-attention
    gemm(tgt, 256, F(B0 + 4), 256, F(B0 + 5), qb, 256, 800, 256, 256, 0);
    gemm(mem, 256, F(B0 + 4) + 256 * 256, 256, F(B0 + 5) + 256,
         kv, 512, 32768, 512, 256, 0);
    attention_kernel<<<dim3(100, 8, 8), 256, 0, stream>>>(
        qb, 256, kv, kv + 256, 512, 4096, attn, 100, scale_att);
    gemm(attn, 256, F(B0 + 6), 256, F(B0 + 7), proj, 256, 800, 256, 256, 0);
    add_ln_kernel<<<800, 256, 0, stream>>>(tgt, proj, F(B0 + 14), F(B0 + 15), tgt);
    // FFN
    gemm(tgt, 256, F(B0 + 8), 256, F(B0 + 9), ff1, 1024, 800, 1024, 256, 1);
    gemm(ff1, 1024, F(B0 + 10), 1024, F(B0 + 11), ff2, 256, 800, 256, 1024, 0);
    add_ln_kernel<<<800, 256, 0, stream>>>(tgt, ff2, F(B0 + 16), F(B0 + 17), tgt);
  }

  // ---- heads ----
  gemm(tgt, 256, F(95), 256, F(96), o_cls, 50, 800, 50, 256, 0);
  softmax50_kernel<<<800, 64, 0, stream>>>(o_cls, o_prob);
  gemm(tgt, 256, F(97), 256, F(98), tmp, 256, 800, 256, 256, 1);
  gemm(tmp, 256, F(99), 256, F(100), o_bbox, 4, 800, 4, 256, 2);
  gemm(tgt, 256, F(101), 256, F(102), tmp, 256, 800, 256, 256, 1);
  gemm(tmp, 256, F(103), 256, F(104), o_lab, 10000, 800, 10000, 256, 0);
  gemm(tgt, 256, F(105), 512, F(106), Ub, 256, 800, 256, 256, 0);
  gemm(tgt, 256, F(105) + 256, 512, nullptr, Vb, 256, 800, 256, 256, 0);
  relations_kernel<<<dim3(9900, 8), 256, 0, stream>>>(Ub, Vb, F(107), F(108), o_rel);
  hipMemcpyAsync(o_dec, tgt, 204800 * sizeof(float),
                 hipMemcpyDeviceToDevice, stream);
}
